// DiffSCM_51410758533236
// MI455X (gfx1250) — matrix operand order repaired and checked by reading, not compiled, not run on hardware
//
#include <hip/hip_runtime.h>
#include <hip/hip_bf16.h>
#include <math.h>

// ---------------------------------------------------------------------------
// DiffSCM fused kernel for MI455X (gfx1250, wave32, WMMA bf16 16x16x32)
//   B = 262144 rows, D = 256 features
//   4 GEMMs fused in one kernel; activations staged as bf16 tiles in LDS.
//   Epilogues use CDNA5 hardware transcendentals (v_exp/v_log/v_rcp).
// ---------------------------------------------------------------------------

#define BROWS 262144
#define DFEAT 256
#define ROWS_PER_WAVE 32     // two 16-row WMMA strips per wave
#define WAVES_PER_BLOCK 2    // 64 threads, 64KB static LDS
#define T_INV (1.0f / 262143.0f)   // 1/(B-1) for linspace(0,1,B)

#define LOG2E 1.4426950408889634f
#define LN2   0.6931471805599453f

typedef __bf16 bf16;
typedef __attribute__((ext_vector_type(16))) __bf16 v16bf;
typedef __attribute__((ext_vector_type(8)))  __bf16 v8bf;
typedef __attribute__((ext_vector_type(4)))  __bf16 v4bf;
typedef __attribute__((ext_vector_type(8)))  float  v8f;

// fast tanh via hardware exp2/rcp: tanh(x) = 1 - 2/(e^{2x}+1)
__device__ __forceinline__ float fast_tanh(float x) {
    float e = __builtin_amdgcn_exp2f(x * (2.0f * LOG2E));
    return 1.0f - 2.0f * __builtin_amdgcn_rcpf(e + 1.0f);
}

// fast softplus via hardware exp2/log2: ln(1+e^v) = ln2*log2(1+2^(v*log2e))
__device__ __forceinline__ float fast_softplus(float v) {
    float sp = LN2 * __builtin_amdgcn_logf(1.0f + __builtin_amdgcn_exp2f(v * LOG2E));
    return (v > 20.0f) ? v : sp;
}

// ---------------------------------------------------------------------------
// Weight swizzle: fp32 row-major W[n][k] (ld = ldw) -> bf16 B-fragment order.
// B-fragment (32x16, K x N) for WMMA bf16: lane l holds column n = l%16,
// K = kt*32 + (l>=16 ? 16 : 0) + j, j = 0..15 contiguous.
// Index: out[(((kt*16 + nt)*32 + lane)*16) + j] = W[n][k].
// ---------------------------------------------------------------------------
__global__ __launch_bounds__(256) void swizzle_weights(const float* __restrict__ W,
                                                       int ldw,
                                                       bf16* __restrict__ out) {
    int idx  = blockIdx.x * 256 + threadIdx.x;     // 65536 total
    int j    = idx & 15;
    int lane = (idx >> 4) & 31;
    int nt   = (idx >> 9) & 15;
    int kt   = (idx >> 13) & 7;
    int n    = nt * 16 + (lane & 15);
    int k    = kt * 32 + ((lane >= 16) ? 16 : 0) + j;
    out[idx] = (bf16)W[(size_t)n * ldw + k];
}

// ---------------------------------------------------------------------------
// Fragment loaders
// ---------------------------------------------------------------------------
__device__ __forceinline__ v16bf load_afrag(const bf16* __restrict__ buf,
                                            int lane, int strip, int kt) {
    // A (16x32 bf16): lane l -> row m=l%16 (+16*strip); K = 16*(l/16) + 0..15 contiguous (same order as the B swizzle)
    int m  = (lane & 15) + strip * 16;
    int hi = lane >> 4;
    const bf16* p = buf + m * DFEAT + kt * 32 + hi * 16;
    v8bf lo = *(const v8bf*)p;          // K = base .. base+7   (ds_load_b128)
    v8bf hb = *(const v8bf*)(p + 8);    // K = base+8..base+15  (ds_load_b128)
    v16bf r;
#pragma unroll
    for (int i = 0; i < 8; ++i) { r[i] = lo[i]; r[i + 8] = hb[i]; }
    return r;
}

__device__ __forceinline__ v16bf load_bfrag(const bf16* __restrict__ Ws,
                                            int lane, int nt, int kt) {
    // pre-swizzled: one contiguous 32-byte load per lane
    return *(const v16bf*)(Ws + ((((kt << 4) + nt) << 5) + lane) * 16);
}

__device__ __forceinline__ void wait_lds() {
    asm volatile("s_wait_dscnt 0x0" ::: "memory");
}

// ---------------------------------------------------------------------------
// Main fused kernel
// ---------------------------------------------------------------------------
__global__ __launch_bounds__(WAVES_PER_BLOCK * 32) void diffscm_fused(
    const float* __restrict__ x,
    const float* __restrict__ noise,
    const float* __restrict__ x0,
    const float* __restrict__ t_rand,
    const float* __restrict__ eps,
    const float* __restrict__ W1raw,   // [256][257] row-major, for t-column
    const float* __restrict__ b1,
    const float* __restrict__ b2,
    const float* __restrict__ bm,
    const float* __restrict__ bv,
    const bf16*  __restrict__ W1s,     // swizzled bf16 B-fragments
    const bf16*  __restrict__ W2s,
    const bf16*  __restrict__ Wms,
    const bf16*  __restrict__ Wvs,
    float* __restrict__ out) {

    // per-wave private LDS: bufA (xt' -> xt), bufB (h); 32 x 256 bf16 each
    __shared__ __align__(16) bf16 smem[WAVES_PER_BLOCK * 2 * ROWS_PER_WAVE * DFEAT];

    const int lane = threadIdx.x & 31;
    const int wave = threadIdx.x >> 5;
    bf16* bufA = smem + wave * (2 * ROWS_PER_WAVE * DFEAT);
    bf16* bufB = bufA + ROWS_PER_WAVE * DFEAT;

    const int row0 = (blockIdx.x * WAVES_PER_BLOCK + wave) * ROWS_PER_WAVE;

    // ------------------------------------------------------------------
    // Phase 1: xt' = x0 + noise * sqrt(1 - t); store bf16 row-major in bufA
    // lane handles rows (lane%16, lane%16+16), cols half (lane/16)*128
    // ------------------------------------------------------------------
    {
        const int rloc  = lane & 15;
        const int chalf = (lane >> 4) * 128;
#pragma unroll
        for (int s = 0; s < 2; ++s) {
            const int r  = s * 16 + rloc;
            const int gm = row0 + r;
            const float t  = (float)gm * T_INV;
            const float sq = sqrtf(1.0f - t);
            const float4* px = (const float4*)(x0    + (size_t)gm * DFEAT + chalf);
            const float4* pn = (const float4*)(noise + (size_t)gm * DFEAT + chalf);
            bf16* dst = bufA + r * DFEAT + chalf;
#pragma unroll 4
            for (int c = 0; c < 32; ++c) {
                float4 a = px[c], nz = pn[c];
                v4bf p;
                p[0] = (bf16)(a.x + nz.x * sq);
                p[1] = (bf16)(a.y + nz.y * sq);
                p[2] = (bf16)(a.z + nz.z * sq);
                p[3] = (bf16)(a.w + nz.w * sq);
                *(v4bf*)(dst + c * 4) = p;
            }
        }
    }
    wait_lds();

    const int hi  = lane >> 4;
    const int nlo = lane & 15;

    // ------------------------------------------------------------------
    // Phase 2: h = relu(xt' @ W1[:, :256].T + t*W1[:,256] + b1) -> bufB
    // ------------------------------------------------------------------
    for (int nt = 0; nt < 16; ++nt) {
        v8f acc0 = {}, acc1 = {};
#pragma unroll
        for (int kt = 0; kt < 8; ++kt) {
            v16bf b  = load_bfrag(W1s, lane, nt, kt);
            v16bf a0 = load_afrag(bufA, lane, 0, kt);
            v16bf a1 = load_afrag(bufA, lane, 1, kt);
            acc0 = __builtin_amdgcn_wmma_f32_16x16x32_bf16(false, a0, false, b,
                                                           (short)0, acc0, false, false);
            acc1 = __builtin_amdgcn_wmma_f32_16x16x32_bf16(false, a1, false, b,
                                                           (short)0, acc1, false, false);
        }
        const int   n    = nt * 16 + nlo;
        const float bias = b1[n];
        const float wt   = W1raw[(size_t)n * 257 + 256];   // t-column of W1
#pragma unroll
        for (int i = 0; i < 8; ++i) {
            int m0 = i + 8 * hi;                 // strip 0 row (local)
            float t0 = (float)(row0 + m0) * T_INV;
            float h0 = fmaxf(acc0[i] + bias + t0 * wt, 0.0f);
            bufB[m0 * DFEAT + n] = (bf16)h0;
            int m1 = m0 + 16;                    // strip 1 row
            float t1 = (float)(row0 + m1) * T_INV;
            float h1 = fmaxf(acc1[i] + bias + t1 * wt, 0.0f);
            bufB[m1 * DFEAT + n] = (bf16)h1;
        }
    }
    wait_lds();

    // ------------------------------------------------------------------
    // Phase 3: ft = tanh(h @ W2.T + b2); xt = xt' + (1-t)*ft -> bufA (in place)
    // ------------------------------------------------------------------
    for (int nt = 0; nt < 16; ++nt) {
        v8f acc0 = {}, acc1 = {};
#pragma unroll
        for (int kt = 0; kt < 8; ++kt) {
            v16bf b  = load_bfrag(W2s, lane, nt, kt);
            v16bf a0 = load_afrag(bufB, lane, 0, kt);
            v16bf a1 = load_afrag(bufB, lane, 1, kt);
            acc0 = __builtin_amdgcn_wmma_f32_16x16x32_bf16(false, a0, false, b,
                                                           (short)0, acc0, false, false);
            acc1 = __builtin_amdgcn_wmma_f32_16x16x32_bf16(false, a1, false, b,
                                                           (short)0, acc1, false, false);
        }
        const int   n    = nt * 16 + nlo;
        const float bias = b2[n];
#pragma unroll
        for (int i = 0; i < 8; ++i) {
            int m0 = i + 8 * hi;
            float t0  = (float)(row0 + m0) * T_INV;
            float ft0 = fast_tanh(acc0[i] + bias);
            float xp0 = (float)bufA[m0 * DFEAT + n];
            bufA[m0 * DFEAT + n] = (bf16)(xp0 + (1.0f - t0) * ft0);
            int m1 = m0 + 16;
            float t1  = (float)(row0 + m1) * T_INV;
            float ft1 = fast_tanh(acc1[i] + bias);
            float xp1 = (float)bufA[m1 * DFEAT + n];
            bufA[m1 * DFEAT + n] = (bf16)(xp1 + (1.0f - t1) * ft1);
        }
    }
    wait_lds();

    // ------------------------------------------------------------------
    // Phase 4: mu = xt@Wm.T + bm ; sigma = softplus(xt@Wv.T + bv)
    //          out = x + t_rand*((mu + sigma*eps) - x)
    // ------------------------------------------------------------------
    // hoist per-row t_rand out of the n-tile loop (16 rows owned per lane)
    float trv[16];
#pragma unroll
    for (int s = 0; s < 2; ++s)
#pragma unroll
        for (int i = 0; i < 8; ++i)
            trv[s * 8 + i] = t_rand[row0 + i + 8 * hi + 16 * s];

    for (int nt = 0; nt < 16; ++nt) {
        v8f am0 = {}, am1 = {}, av0 = {}, av1 = {};
#pragma unroll
        for (int kt = 0; kt < 8; ++kt) {
            v16bf bmf = load_bfrag(Wms, lane, nt, kt);
            v16bf bvf = load_bfrag(Wvs, lane, nt, kt);
            v16bf a0  = load_afrag(bufA, lane, 0, kt);
            v16bf a1  = load_afrag(bufA, lane, 1, kt);
            am0 = __builtin_amdgcn_wmma_f32_16x16x32_bf16(false, a0, false, bmf,
                                                          (short)0, am0, false, false);
            am1 = __builtin_amdgcn_wmma_f32_16x16x32_bf16(false, a1, false, bmf,
                                                          (short)0, am1, false, false);
            av0 = __builtin_amdgcn_wmma_f32_16x16x32_bf16(false, a0, false, bvf,
                                                          (short)0, av0, false, false);
            av1 = __builtin_amdgcn_wmma_f32_16x16x32_bf16(false, a1, false, bvf,
                                                          (short)0, av1, false, false);
        }
        const int   n   = nt * 16 + nlo;
        const float bmv = bm[n];
        const float bvv = bv[n];
#pragma unroll
        for (int i = 0; i < 8; ++i) {
#pragma unroll
            for (int s = 0; s < 2; ++s) {
                int m  = i + 8 * hi + 16 * s;
                int gm = row0 + m;
                size_t off = (size_t)gm * DFEAT + n;
                float mu  = (s ? am1[i] : am0[i]) + bmv;
                float sv  = (s ? av1[i] : av0[i]) + bvv;
                float sig = fast_softplus(sv);
                float xprime = mu + sig * eps[off];
                float xv = x[off];
                // (1-tr)*x + tr*x' == x + tr*(x' - x)
                out[off] = fmaf(trv[s * 8 + i], xprime - xv, xv);
            }
        }
    }
}

// ---------------------------------------------------------------------------
// Launch
// ---------------------------------------------------------------------------
extern "C" void kernel_launch(void* const* d_in, const int* in_sizes, int n_in,
                              void* d_out, int out_size, void* d_ws, size_t ws_size,
                              hipStream_t stream) {
    (void)in_sizes; (void)n_in; (void)out_size; (void)ws_size;

    const float* x      = (const float*)d_in[0];
    const float* noise  = (const float*)d_in[1];
    const float* x0     = (const float*)d_in[2];
    const float* t_rand = (const float*)d_in[3];
    const float* eps    = (const float*)d_in[4];
    const float* W1     = (const float*)d_in[5];
    const float* b1     = (const float*)d_in[6];
    const float* W2     = (const float*)d_in[7];
    const float* b2     = (const float*)d_in[8];
    const float* Wm     = (const float*)d_in[9];
    const float* bm     = (const float*)d_in[10];
    const float* Wv     = (const float*)d_in[11];
    const float* bv     = (const float*)d_in[12];
    float* out = (float*)d_out;

    // workspace: 4 swizzled bf16 weight matrices, 128KB each
    bf16* W1s = (bf16*)d_ws;
    bf16* W2s = W1s + 65536;
    bf16* Wms = W2s + 65536;
    bf16* Wvs = Wms + 65536;

    swizzle_weights<<<256, 256, 0, stream>>>(W1, 257, W1s);
    swizzle_weights<<<256, 256, 0, stream>>>(W2, 256, W2s);
    swizzle_weights<<<256, 256, 0, stream>>>(Wm, 256, Wms);
    swizzle_weights<<<256, 256, 0, stream>>>(Wv, 256, Wvs);

    const int blocks = BROWS / (WAVES_PER_BLOCK * ROWS_PER_WAVE);   // 4096
    diffscm_fused<<<blocks, WAVES_PER_BLOCK * 32, 0, stream>>>(
        x, noise, x0, t_rand, eps, W1, b1, b2, bm, bv,
        W1s, W2s, Wms, Wvs, out);
}
